// Graph2Mesh_59889023975748
// MI455X (gfx1250) — matrix operand order repaired and checked by reading, not compiled, not run on hardware
//
#include <hip/hip_runtime.h>

typedef __attribute__((ext_vector_type(16))) __bf16 v16bf;
typedef __attribute__((ext_vector_type(8)))  float  v8f;

#define NN 100000      // nodes
#define NE 3200000     // edges
#define NG 64          // graphs

// ---------------- utility kernels ----------------

__global__ void zero_f32(float* p, int n) {
    int i = blockIdx.x * blockDim.x + threadIdx.x;
    if (i < n) p[i] = 0.0f;
}

__global__ void deg_accum(const int* __restrict__ dst, float* deg, int E) {
    int e = blockIdx.x * blockDim.x + threadIdx.x;
    if (e < E) unsafeAtomicAdd(&deg[dst[e]], 1.0f);
}

__global__ void finalize_dinv(float* d, int n) {
    int i = blockIdx.x * blockDim.x + threadIdx.x;
    if (i < n) d[i] = rsqrtf(d[i] + 1.0f);   // +1 self loop, deg >= 1
}

// out[n,f] = h[n,f] * dinv[n]^2   (self-loop contribution of S = D^-1/2(A+I)D^-1/2)
__global__ void selfloop_scale(const float* __restrict__ h, const float* __restrict__ dinv,
                               float* out, int N, int F) {
    int i = blockIdx.x * blockDim.x + threadIdx.x;
    if (i < N * F) {
        int n = i / F;
        float s = dinv[n];
        out[i] = h[i] * s * s;
    }
}

// agg[dst] += h[src] * dinv[src]*dinv[dst], F = 64 (16 threads/edge, float4 each)
__global__ void edge_scatter64(const float* __restrict__ h, const int* __restrict__ src,
                               const int* __restrict__ dst, const float* __restrict__ dinv,
                               float* agg, int E) {
    long long t = (long long)blockIdx.x * blockDim.x + threadIdx.x;
    int e = (int)(t >> 4);
    if (e >= E) return;
    int fo = ((int)t & 15) << 2;
    int s = src[e], d = dst[e];
    float nm = dinv[s] * dinv[d];
    const float4 hv = *reinterpret_cast<const float4*>(h + (size_t)s * 64 + fo);
    float* ap = agg + (size_t)d * 64 + fo;
    unsafeAtomicAdd(ap + 0, hv.x * nm);
    unsafeAtomicAdd(ap + 1, hv.y * nm);
    unsafeAtomicAdd(ap + 2, hv.z * nm);
    unsafeAtomicAdd(ap + 3, hv.w * nm);
}

__global__ void bias_relu(float* a, const float* __restrict__ b, int N, int F) {
    int i = blockIdx.x * blockDim.x + threadIdx.x;
    if (i < N * F) a[i] = fmaxf(a[i] + b[i % F], 0.0f);
}

// ---------------- weight repack: B[K,N] f32 -> fragment-major bf16 ----------------
// Fragment order: contiguous k per lane half, same as the GEMM's A fragment: for tile (nt, kt),
// lane L holds b[i] = B[kt*32 + (L>>4)*16 + i][nt*16 + (L&15)], i = 0..15.
// Packed address: P[((nt*KT + kt)*32 + L)*16 + i]  -> one 32B contiguous run/lane.
__global__ void pack_b_bf16(const float* __restrict__ B, __bf16* __restrict__ P,
                            int K, int N) {
    int t = blockIdx.x * blockDim.x + threadIdx.x;
    int KT = K >> 5;
    int total = (N >> 4) * KT * 32;
    if (t >= total) return;
    int lane = t & 31;
    int g    = t >> 5;          // nt*KT + kt
    int kt   = g % KT;
    int half = lane >> 4, l15 = lane & 15;
    int nt   = g / KT;
    const float* src = B + (size_t)(kt * 32 + half * 16) * N + nt * 16 + l15;
    __bf16* dst = P + ((size_t)g * 32 + lane) * 16;
    #pragma unroll
    for (int i = 0; i < 16; ++i) dst[i] = (__bf16)src[(size_t)i * N];
}

// ---------------- WMMA GEMM: C[M,N] = act(A[M,K] @ B[K,N] + bias) ----------------
// A: f32 row-major (converted inline); B: pre-packed fragment-major bf16.
// block = (32,2): each wave owns a 16x32 output tile (two WMMA tiles sharing one
// A fragment). Requires M%32==0, N%32==0, K%32==0. No divergence -> EXEC all-ones.
template<int K, bool BIAS, bool RELU>
__global__ __launch_bounds__(64) void gemm_bf16_wmma(
        const float* __restrict__ A, const __bf16* __restrict__ Bp,
        const float* __restrict__ bias, float* __restrict__ C,
        int M, int N) {
    constexpr int KT = K >> 5;
    const int lane  = threadIdx.x;                 // 0..31
    const int mtile = blockIdx.x * 2 + threadIdx.y;
    const int half  = lane >> 4;                   // lane group 0/1
    const int l15   = lane & 15;
    const int mrow  = (mtile << 4) + l15;
    const int nt0   = blockIdx.y * 2;
    const int ncol0 = (nt0 << 4) + l15;
    const int ncol1 = ncol0 + 16;

    const v16bf* B0 = reinterpret_cast<const v16bf*>(Bp) + (size_t)nt0 * KT * 32 + lane;
    const v16bf* B1 = B0 + (size_t)KT * 32;

    v8f c0 = {}, c1 = {};
    #pragma unroll 4
    for (int kt = 0; kt < KT; ++kt) {
        v16bf a;
        // A fragment (16x32, row = lane%16): half 0 -> K {0..15}, half 1 -> K {16..31}
        const float* arow = A + (size_t)mrow * K + kt * 32 + half * 16;
        #pragma unroll
        for (int i = 0; i < 8; ++i) a[i]     = (__bf16)arow[i];
        #pragma unroll
        for (int i = 0; i < 8; ++i) a[8 + i] = (__bf16)arow[8 + i];
        // B fragments: contiguous 32B per lane from the packed buffer
        v16bf b0 = B0[(size_t)kt * 32];
        v16bf b1 = B1[(size_t)kt * 32];

        c0 = __builtin_amdgcn_wmma_f32_16x16x32_bf16(
                 false, a, false, b0, (short)0, c0, false, false);
        c1 = __builtin_amdgcn_wmma_f32_16x16x32_bf16(
                 false, a, false, b1, (short)0, c1, false, false);
    }

    float bv0 = 0.0f, bv1 = 0.0f;
    if constexpr (BIAS) { bv0 = bias[ncol0]; bv1 = bias[ncol1]; }
    // C/D layout: VGPR r -> M = half*8 + r, N = lane%16
    #pragma unroll
    for (int r = 0; r < 8; ++r) {
        int mo = (mtile << 4) + (half << 3) + r;
        float v0 = c0[r] + bv0;
        float v1 = c1[r] + bv1;
        if constexpr (RELU) { v0 = fmaxf(v0, 0.0f); v1 = fmaxf(v1, 0.0f); }
        C[(size_t)mo * N + ncol0] = v0;
        C[(size_t)mo * N + ncol1] = v1;
    }
}

// ---------------- mean pool over graphs ----------------
__global__ void pool_accum(const float* __restrict__ h, const int* __restrict__ batch,
                           float* sums, float* cnt, int N) {
    long long t = (long long)blockIdx.x * blockDim.x + threadIdx.x;
    int n = (int)(t >> 5);                 // 32 threads per node (128 feats / 4)
    if (n >= N) return;
    int fo = ((int)t & 31) << 2;
    int g = batch[n];
    const float4 hv = *reinterpret_cast<const float4*>(h + (size_t)n * 128 + fo);
    float* sp = sums + (size_t)g * 128 + fo;
    unsafeAtomicAdd(sp + 0, hv.x);
    unsafeAtomicAdd(sp + 1, hv.y);
    unsafeAtomicAdd(sp + 2, hv.z);
    unsafeAtomicAdd(sp + 3, hv.w);
    if (fo == 0) unsafeAtomicAdd(&cnt[g], 1.0f);
}

__global__ void pool_final(const float* __restrict__ sums, const float* __restrict__ cnt,
                           float* g) {
    int i = blockIdx.x * blockDim.x + threadIdx.x;
    if (i < NG * 128) g[i] = sums[i] / fmaxf(cnt[i >> 7], 1.0f);
}

// ---------------- scalar GEMM for final 512 -> 100 layer ----------------
__global__ void gemm_naive(const float* __restrict__ A, const float* __restrict__ B,
                           const float* __restrict__ bias, float* __restrict__ C,
                           int M, int K, int N) {
    int idx = blockIdx.x * blockDim.x + threadIdx.x;
    if (idx >= M * N) return;
    int m = idx / N, n = idx % N;
    float acc = bias[n];
    for (int k = 0; k < K; ++k) acc = fmaf(A[m * K + k], B[k * N + n], acc);
    C[idx] = acc;
}

// ---------------- host side ----------------
extern "C" void kernel_launch(void* const* d_in, const int* in_sizes, int n_in,
                              void* d_out, int out_size, void* d_ws, size_t ws_size,
                              hipStream_t stream) {
    const float* x   = (const float*)d_in[0];
    const int*   ei  = (const int*)d_in[1];
    const int*   src = ei;
    const int*   dst = ei + NE;
    const int*   batch = (const int*)d_in[2];
    const float* W1 = (const float*)d_in[3];  const float* b1  = (const float*)d_in[4];
    const float* W2 = (const float*)d_in[5];  const float* b2  = (const float*)d_in[6];
    const float* Wf1= (const float*)d_in[7];  const float* bf1 = (const float*)d_in[8];
    const float* Wf2= (const float*)d_in[9];  const float* bf2 = (const float*)d_in[10];
    const float* Wf3= (const float*)d_in[11]; const float* bf3 = (const float*)d_in[12];
    const float* Wf4= (const float*)d_in[13]; const float* bf4 = (const float*)d_in[14];
    float* out = (float*)d_out;

    float* ws   = (float*)d_ws;
    float* dinv = ws;                               // 100000
    float* h1   = ws + 131072;                      // N*64
    float* agg1 = h1   + (size_t)NN * 64;           // N*64 (becomes relu'd h1)
    float* p    = agg1 + (size_t)NN * 64;           // N*64 (S * h1relu)
    float* h2   = p    + (size_t)NN * 64;           // N*128
    float* sums = h2   + (size_t)NN * 128;          // 64*128
    float* cnt  = sums + NG * 128;                  // 64
    float* g0   = cnt  + NG;                        // 64*128
    float* g1   = g0   + NG * 128;                  // 64*256
    float* g2   = g1   + NG * 256;                  // 64*512
    float* g3   = g2   + NG * 512;                  // 64*512
    // packed bf16 weights (32B-aligned; offsets are multiples of 1KB)
    __bf16* pW1  = (__bf16*)(g3 + NG * 512);        // 128*64
    __bf16* pW2  = pW1  + 128 * 64;                 // 64*128
    __bf16* pWf1 = pW2  + 64 * 128;                 // 128*256
    __bf16* pWf2 = pWf1 + 128 * 256;                // 256*512
    __bf16* pWf3 = pWf2 + 256 * 512;                // 512*512

    const int T = 256;
    dim3 wb(32, 2);  // WMMA block: 2 waves, each owning a 16x32 tile

    // one-time (per launch) weight repacks to fragment-major bf16
    pack_b_bf16<<<(4  * 4  * 32 + T - 1) / T, T, 0, stream>>>(W1,  pW1,  128, 64);
    pack_b_bf16<<<(8  * 2  * 32 + T - 1) / T, T, 0, stream>>>(W2,  pW2,  64,  128);
    pack_b_bf16<<<(16 * 4  * 32 + T - 1) / T, T, 0, stream>>>(Wf1, pWf1, 128, 256);
    pack_b_bf16<<<(32 * 8  * 32 + T - 1) / T, T, 0, stream>>>(Wf2, pWf2, 256, 512);
    pack_b_bf16<<<(32 * 16 * 32 + T - 1) / T, T, 0, stream>>>(Wf3, pWf3, 512, 512);

    // degrees -> dinv (in place)
    zero_f32<<<(NN + T - 1) / T, T, 0, stream>>>(dinv, NN);
    zero_f32<<<(NG * 128 + NG + T - 1) / T, T, 0, stream>>>(sums, NG * 128 + NG);
    deg_accum<<<(NE + T - 1) / T, T, 0, stream>>>(dst, dinv, NE);
    finalize_dinv<<<(NN + T - 1) / T, T, 0, stream>>>(dinv, NN);

    // layer 1: h1 = x @ W1, agg1 = S*h1, relu(+b1)
    gemm_bf16_wmma<128, false, false>
        <<<dim3(NN / 32, 64 / 32), wb, 0, stream>>>(x, pW1, nullptr, h1, NN, 64);
    selfloop_scale<<<(NN * 64 + T - 1) / T, T, 0, stream>>>(h1, dinv, agg1, NN, 64);
    edge_scatter64<<<(int)(((long long)NE * 16 + T - 1) / T), T, 0, stream>>>(
        h1, src, dst, dinv, agg1, NE);
    bias_relu<<<(NN * 64 + T - 1) / T, T, 0, stream>>>(agg1, b1, NN, 64);

    // layer 2 (reassociated): p = S*agg1 (64 feats), h2 = relu(p @ W2 + b2)
    selfloop_scale<<<(NN * 64 + T - 1) / T, T, 0, stream>>>(agg1, dinv, p, NN, 64);
    edge_scatter64<<<(int)(((long long)NE * 16 + T - 1) / T), T, 0, stream>>>(
        agg1, src, dst, dinv, p, NE);
    gemm_bf16_wmma<64, true, true>
        <<<dim3(NN / 32, 128 / 32), wb, 0, stream>>>(p, pW2, b2, h2, NN, 128);

    // global mean pool -> g0 [64,128]
    pool_accum<<<(int)(((long long)NN * 32 + T - 1) / T), T, 0, stream>>>(
        h2, batch, sums, cnt, NN);
    pool_final<<<(NG * 128 + T - 1) / T, T, 0, stream>>>(sums, cnt, g0);

    // MLP head (WMMA for the three relu layers)
    gemm_bf16_wmma<128, true, true>
        <<<dim3(NG / 32, 256 / 32), wb, 0, stream>>>(g0, pWf1, bf1, g1, NG, 256);
    gemm_bf16_wmma<256, true, true>
        <<<dim3(NG / 32, 512 / 32), wb, 0, stream>>>(g1, pWf2, bf2, g2, NG, 512);
    gemm_bf16_wmma<512, true, true>
        <<<dim3(NG / 32, 512 / 32), wb, 0, stream>>>(g2, pWf3, bf3, g3, NG, 512);
    gemm_naive<<<(NG * 100 + T - 1) / T, T, 0, stream>>>(g3, Wf4, bf4, out,
                                                         NG, 512, 100);
}
